// SAGE_sup_55009941127683
// MI455X (gfx1250) — matrix operand order repaired and checked by reading, not compiled, not run on hardware
//
#include <hip/hip_runtime.h>
#include <hip/hip_bf16.h>
#include <math.h>

// ---------------------------------------------------------------------------
// GraphSAGE 2-layer forward for MI455X (gfx1250, wave32, WMMA).
//
//   h   = relu( scatter_mean(x[src]->dst) @ W1_l + x @ W1_r + b1 )
//   out = sigmoid( scatter_mean(h[src]->dst) @ W2_l + h @ W2_r + b2 )
//
// Layer-2 uses transform-before-aggregate (mean is linear): scatter width
// 100 instead of 256 channels (64M instead of 164M f32 atomics).
//
// GEMMs: v_wmma_f32_16x16x32_bf16, f32 accumulate.
//   - Weights pre-transposed+converted to bf16 [n][k] so B frags are two 16B
//     contiguous runs per lane (2x global_load_b128).
//   - Activations staged in LDS as bf16 so A frags are 2x ds_load_b128.
//   - 32 rows per block: each B frag feeds 2 WMMAs.
// ---------------------------------------------------------------------------

#define N_NODES 100000
#define N_EDGES 640000
#define IN_CH   128
#define HID_CH  256
#define CLS_CH  100
#define CLS_PAD 128   // padded class stride (power of 2 for cheap idx math)
#define CLS_T   112   // 7 x 16 column tiles for layer-2 GEMMs

typedef __attribute__((ext_vector_type(16))) __bf16 v16bf;
typedef __attribute__((ext_vector_type(8)))  __bf16 v8bf;
typedef __attribute__((ext_vector_type(4)))  __bf16 v4bf;
typedef __attribute__((ext_vector_type(8)))  float  v8f;

__device__ __forceinline__ v8f wmma_bf16(v16bf a, v16bf b, v8f c) {
  // (neg_a, A, neg_b, B, c_mod, C, reuse_a, reuse_b)
  return __builtin_amdgcn_wmma_f32_16x16x32_bf16(false, a, false, b,
                                                 (short)0, c, false, false);
}

__device__ __forceinline__ v16bf cat8(v8bf lo, v8bf hi) {
  return __builtin_shufflevector(lo, hi, 0, 1, 2, 3, 4, 5, 6, 7,
                                 8, 9, 10, 11, 12, 13, 14, 15);
}

// ---------------------------------------------------------------------------
// Weight prep: transpose + convert to bf16, pad layer-2 cols to 112.
//   w1T[n*128 + k] = W1[k*256 + n]          (n < 256, k < 128)
//   w2T[n*256 + k] = n<100 ? W2[k*100 + n]  (n < 112, k < 256)
// ---------------------------------------------------------------------------
#define W1T_ELEMS (HID_CH * IN_CH)   // 32768
#define W2T_ELEMS (CLS_T * HID_CH)   // 28672
#define PREP_TOTAL (2 * W1T_ELEMS + 2 * W2T_ELEMS)   // 122880

__global__ __launch_bounds__(256) void sage_prep_weights(
    const float* __restrict__ W1l, const float* __restrict__ W1r,
    const float* __restrict__ W2l, const float* __restrict__ W2r,
    __bf16* __restrict__ w1lT, __bf16* __restrict__ w1rT,
    __bf16* __restrict__ w2lT, __bf16* __restrict__ w2rT) {
  int idx = blockIdx.x * 256 + threadIdx.x;
  if (idx < 2 * W1T_ELEMS) {
    int m = idx;
    const float* W = W1l; __bf16* T = w1lT;
    if (m >= W1T_ELEMS) { m -= W1T_ELEMS; W = W1r; T = w1rT; }
    int n = m >> 7, k = m & 127;
    T[m] = (__bf16)W[k * HID_CH + n];
  } else if (idx < PREP_TOTAL) {
    int m = idx - 2 * W1T_ELEMS;
    const float* W = W2l; __bf16* T = w2lT;
    if (m >= W2T_ELEMS) { m -= W2T_ELEMS; W = W2r; T = w2rT; }
    int n = m >> 8, k = m & 255;
    T[m] = (n < CLS_CH) ? (__bf16)W[k * CLS_CH + n] : (__bf16)0.0f;
  }
}

// ---------------------------------------------------------------------------
// Scatter 1: agg1[dst] += x[src] (128 ch), deg[dst] += 1.
// Wave = 32 consecutive channels of one edge -> 128B coalesced loads and
// 128B atomic segments (all L2-resident).
// ---------------------------------------------------------------------------
__global__ __launch_bounds__(256) void sage_scatter1(
    const float* __restrict__ x, const int* __restrict__ src,
    const int* __restrict__ dst, float* __restrict__ agg1,
    float* __restrict__ deg) {
  unsigned idx = blockIdx.x * 256u + threadIdx.x;   // grid sized exactly E*128
  unsigned e = idx >> 7, c = idx & 127u;
  int s = src[e], d = dst[e];
  atomicAdd(&agg1[(unsigned)d * (unsigned)IN_CH + c],
            x[(unsigned)s * (unsigned)IN_CH + c]);
  if (c == 0u) atomicAdd(&deg[d], 1.0f);
}

// ---------------------------------------------------------------------------
// Scatter 2: acc2[dst, c] += t[src, c] for c < 100 (stride CLS_PAD).
// ---------------------------------------------------------------------------
__global__ __launch_bounds__(256) void sage_scatter2(
    const float* __restrict__ t, const int* __restrict__ src,
    const int* __restrict__ dst, float* __restrict__ acc2) {
  unsigned idx = blockIdx.x * 256u + threadIdx.x;   // grid sized exactly E*128
  unsigned e = idx >> 7, c = idx & 127u;
  if (c < (unsigned)CLS_CH) {
    int s = src[e], d = dst[e];
    atomicAdd(&acc2[(unsigned)d * (unsigned)CLS_PAD + c],
              t[(unsigned)s * (unsigned)CLS_PAD + c]);
  }
}

// ---------------------------------------------------------------------------
// GEMM1: h = relu(mean @ W1_l + x @ W1_r + b1)   [100000x128]@[128x256]
// Block: 32 rows, 8 waves; wave owns 32 cols x 32 rows (4 accumulators).
// LDS: mean,x as bf16 [32][136] (stride 136: 272B rows keep 16B alignment
// and spread banks).
// ---------------------------------------------------------------------------
#define S1_STRIDE 136

__global__ __launch_bounds__(256) void sage_gemm1(
    const float* __restrict__ x, const float* __restrict__ agg1,
    const float* __restrict__ deg, const __bf16* __restrict__ WlT,
    const __bf16* __restrict__ WrT, const float* __restrict__ bias,
    float* __restrict__ h) {
  __shared__ __bf16 sA[2 * 32 * S1_STRIDE];   // [0]: mean, [1]: x
  const int row0 = blockIdx.x << 5;
  const int tid  = threadIdx.x;

  // Stage 32 rows x 128 ch of mean and x, f32 -> bf16, 4 ch per thread.
  for (int idx = tid; idx < 32 * 32; idx += 256) {
    int r = idx >> 5, q = (idx & 31) << 2;
    int g = row0 + r;
    float d = deg[g]; d = (d < 1.0f) ? 1.0f : d;
    float inv = 1.0f / d;
    float4 av = *(const float4*)&agg1[g * IN_CH + q];
    float4 xv = *(const float4*)&x[g * IN_CH + q];
    v4bf mv = {(__bf16)(av.x * inv), (__bf16)(av.y * inv),
               (__bf16)(av.z * inv), (__bf16)(av.w * inv)};
    v4bf xb = {(__bf16)xv.x, (__bf16)xv.y, (__bf16)xv.z, (__bf16)xv.w};
    *(v4bf*)&sA[r * S1_STRIDE + q]                  = mv;
    *(v4bf*)&sA[32 * S1_STRIDE + r * S1_STRIDE + q] = xb;
  }
  __syncthreads();

  const int lane = tid & 31;
  const int n0   = (tid >> 5) << 5;            // 32-col slab per wave
  const int m    = lane & 15;
  const int kb   = (lane >> 4) << 3;           // A-frag K-half offset (0/8)
  const int n    = lane & 15;
  const int kh   = (lane >> 4) << 3;           // B-frag K-half offset (0/8)

  v8f acc[2][2] = {};   // [row tile][col tile]
  for (int pass = 0; pass < 2; ++pass) {
    const __bf16* WT    = pass ? WrT : WlT;
    const __bf16* sbase = &sA[pass * 32 * S1_STRIDE];
    const __bf16* wrow0 = WT + (n0 + n) * IN_CH;
    const __bf16* wrow1 = WT + (n0 + 16 + n) * IN_CH;
#pragma unroll
    for (int kt = 0; kt < 4; ++kt) {           // K = 128 = 4 x 32
      const int k0 = (kt << 5) + kb;
      // A frags (rows m, m+16): two contiguous 8-K chunks each.
      v16bf a0 = cat8(*(const v8bf*)&sbase[m * S1_STRIDE + k0],
                      *(const v8bf*)&sbase[m * S1_STRIDE + k0 + 16]);
      v16bf a1 = cat8(*(const v8bf*)&sbase[(m + 16) * S1_STRIDE + k0],
                      *(const v8bf*)&sbase[(m + 16) * S1_STRIDE + k0 + 16]);
      // B frags: two contiguous 8-K chunks per lane from transposed weights.
      v16bf b0 = cat8(*(const v8bf*)&wrow0[(kt << 5) + kh],
                      *(const v8bf*)&wrow0[(kt << 5) + kh + 16]);
      v16bf b1 = cat8(*(const v8bf*)&wrow1[(kt << 5) + kh],
                      *(const v8bf*)&wrow1[(kt << 5) + kh + 16]);
      acc[0][0] = wmma_bf16(a0, b0, acc[0][0]);
      acc[0][1] = wmma_bf16(a0, b1, acc[0][1]);
      acc[1][0] = wmma_bf16(a1, b0, acc[1][0]);
      acc[1][1] = wmma_bf16(a1, b1, acc[1][1]);
    }
  }

  const int mb  = (lane >> 4) << 3;
  const float bi0 = bias[n0 + n], bi1 = bias[n0 + 16 + n];
#pragma unroll
  for (int rt = 0; rt < 2; ++rt) {
#pragma unroll
    for (int r = 0; r < 8; ++r) {
      float v0 = acc[rt][0][r] + bi0; v0 = (v0 > 0.0f) ? v0 : 0.0f;
      float v1 = acc[rt][1][r] + bi1; v1 = (v1 > 0.0f) ? v1 : 0.0f;
      const int g = row0 + (rt << 4) + mb + r;
      h[g * HID_CH + n0 + n]      = v0;
      h[g * HID_CH + n0 + 16 + n] = v1;
    }
  }
}

// ---------------------------------------------------------------------------
// Layer-2 GEMM core: 32 rows/block, 7 waves (one 16-col tile each), K=256.
// LDS: h rows as bf16 [32][264].
// ---------------------------------------------------------------------------
#define S2_STRIDE 264

// GEMM2a: t = h @ W2_l   (cols padded to 112, stored stride CLS_PAD)
__global__ __launch_bounds__(224) void sage_gemm2a(
    const float* __restrict__ h, const __bf16* __restrict__ WT,
    float* __restrict__ t) {
  __shared__ __bf16 sA[32 * S2_STRIDE];
  const int row0 = blockIdx.x << 5;
  const int tid  = threadIdx.x;

  for (int idx = tid; idx < 32 * 64; idx += 224) {
    int r = idx >> 6, q = (idx & 63) << 2;
    float4 hv = *(const float4*)&h[(row0 + r) * HID_CH + q];
    v4bf hb = {(__bf16)hv.x, (__bf16)hv.y, (__bf16)hv.z, (__bf16)hv.w};
    *(v4bf*)&sA[r * S2_STRIDE + q] = hb;
  }
  __syncthreads();

  const int lane = tid & 31;
  const int n0   = (tid >> 5) << 4;
  const int m    = lane & 15;
  const int kb   = (lane >> 4) << 3;
  const int n    = lane & 15;
  const int kh   = (lane >> 4) << 3;
  const __bf16* wrow = WT + (n0 + n) * HID_CH;

  v8f acc[2] = {};
#pragma unroll
  for (int kt = 0; kt < 8; ++kt) {             // K = 256 = 8 x 32
    const int k0 = (kt << 5) + kb;
    v16bf a0 = cat8(*(const v8bf*)&sA[m * S2_STRIDE + k0],
                    *(const v8bf*)&sA[m * S2_STRIDE + k0 + 16]);
    v16bf a1 = cat8(*(const v8bf*)&sA[(m + 16) * S2_STRIDE + k0],
                    *(const v8bf*)&sA[(m + 16) * S2_STRIDE + k0 + 16]);
    v16bf b0 = cat8(*(const v8bf*)&wrow[(kt << 5) + kh],
                    *(const v8bf*)&wrow[(kt << 5) + kh + 16]);
    acc[0] = wmma_bf16(a0, b0, acc[0]);
    acc[1] = wmma_bf16(a1, b0, acc[1]);
  }

  const int mb = (lane >> 4) << 3;
#pragma unroll
  for (int rt = 0; rt < 2; ++rt)
#pragma unroll
    for (int r = 0; r < 8; ++r)
      t[(row0 + (rt << 4) + mb + r) * CLS_PAD + n0 + n] = acc[rt][r];
}

// GEMM2b (fused epilogue): out = sigmoid(h @ W2_r + b2 + acc2/deg)
__global__ __launch_bounds__(224) void sage_gemm2b(
    const float* __restrict__ h, const __bf16* __restrict__ WT,
    const float* __restrict__ bias, const float* __restrict__ acc2,
    const float* __restrict__ deg, float* __restrict__ out) {
  __shared__ __bf16 sA[32 * S2_STRIDE];
  const int row0 = blockIdx.x << 5;
  const int tid  = threadIdx.x;

  for (int idx = tid; idx < 32 * 64; idx += 224) {
    int r = idx >> 6, q = (idx & 63) << 2;
    float4 hv = *(const float4*)&h[(row0 + r) * HID_CH + q];
    v4bf hb = {(__bf16)hv.x, (__bf16)hv.y, (__bf16)hv.z, (__bf16)hv.w};
    *(v4bf*)&sA[r * S2_STRIDE + q] = hb;
  }
  __syncthreads();

  const int lane = tid & 31;
  const int n0   = (tid >> 5) << 4;
  const int m    = lane & 15;
  const int kb   = (lane >> 4) << 3;
  const int n    = lane & 15;
  const int kh   = (lane >> 4) << 3;
  const int col  = n0 + n;
  const __bf16* wrow = WT + col * HID_CH;

  v8f acc[2] = {};
#pragma unroll
  for (int kt = 0; kt < 8; ++kt) {
    const int k0 = (kt << 5) + kb;
    v16bf a0 = cat8(*(const v8bf*)&sA[m * S2_STRIDE + k0],
                    *(const v8bf*)&sA[m * S2_STRIDE + k0 + 16]);
    v16bf a1 = cat8(*(const v8bf*)&sA[(m + 16) * S2_STRIDE + k0],
                    *(const v8bf*)&sA[(m + 16) * S2_STRIDE + k0 + 16]);
    v16bf b0 = cat8(*(const v8bf*)&wrow[(kt << 5) + kh],
                    *(const v8bf*)&wrow[(kt << 5) + kh + 16]);
    acc[0] = wmma_bf16(a0, b0, acc[0]);
    acc[1] = wmma_bf16(a1, b0, acc[1]);
  }

  const int mb = (lane >> 4) << 3;
  const float bi = (col < CLS_CH) ? bias[col] : 0.0f;
#pragma unroll
  for (int rt = 0; rt < 2; ++rt) {
#pragma unroll
    for (int r = 0; r < 8; ++r) {
      const int g = row0 + (rt << 4) + mb + r;
      float d = deg[g]; d = (d < 1.0f) ? 1.0f : d;
      float u = acc[rt][r] + bi + acc2[g * CLS_PAD + col] * (1.0f / d);
      float s = 1.0f / (1.0f + __expf(-u));
      if (col < CLS_CH) out[g * CLS_CH + col] = s;
    }
  }
}

// ---------------------------------------------------------------------------
// Launch: memsets + 6 kernels, all stream-ordered (graph-capture safe).
// Workspace: deg[N] | agg1[N*128] (reused as acc2) | h[N*256] | t[N*128]
//            | bf16 weights (~246 KB).  Total ~205.5 MB.
// ---------------------------------------------------------------------------
extern "C" void kernel_launch(void* const* d_in, const int* in_sizes, int n_in,
                              void* d_out, int out_size, void* d_ws, size_t ws_size,
                              hipStream_t stream) {
  (void)in_sizes; (void)n_in; (void)out_size; (void)ws_size;

  const float* x   = (const float*)d_in[0];
  const int*   ei  = (const int*)d_in[1];      // [2, E] int32
  const float* W1l = (const float*)d_in[2];
  const float* W1r = (const float*)d_in[3];
  const float* b1  = (const float*)d_in[4];
  const float* W2l = (const float*)d_in[5];
  const float* W2r = (const float*)d_in[6];
  const float* b2  = (const float*)d_in[7];
  const int* src = ei;
  const int* dst = ei + N_EDGES;

  float* ws   = (float*)d_ws;
  float* deg  = ws;                                  // N
  float* agg1 = deg + N_NODES;                       // N*128
  float* h    = agg1 + (size_t)N_NODES * IN_CH;      // N*256
  float* t    = h + (size_t)N_NODES * HID_CH;        // N*128 (CLS_PAD)
  float* acc2 = agg1;                                // alias: agg1 dead after gemm1
  __bf16* w1lT = (__bf16*)(t + (size_t)N_NODES * CLS_PAD);
  __bf16* w1rT = w1lT + W1T_ELEMS;
  __bf16* w2lT = w1rT + W1T_ELEMS;
  __bf16* w2rT = w2lT + W2T_ELEMS;

  sage_prep_weights<<<(PREP_TOTAL + 255) / 256, 256, 0, stream>>>(
      W1l, W1r, W2l, W2r, w1lT, w1rT, w2lT, w2rT);

  // zero deg + agg1 (contiguous)
  hipMemsetAsync(deg, 0,
                 (size_t)(N_NODES + (size_t)N_NODES * IN_CH) * sizeof(float),
                 stream);

  const int scat_blocks = (N_EDGES * 128) / 256;     // 320000
  sage_scatter1<<<scat_blocks, 256, 0, stream>>>(x, src, dst, agg1, deg);

  sage_gemm1<<<N_NODES / 32, 256, 0, stream>>>(x, agg1, deg, w1lT, w1rT, b1, h);

  sage_gemm2a<<<N_NODES / 32, 224, 0, stream>>>(h, w2lT, t);

  hipMemsetAsync(acc2, 0, (size_t)N_NODES * CLS_PAD * sizeof(float), stream);

  sage_scatter2<<<scat_blocks, 256, 0, stream>>>(t, src, dst, acc2);

  sage_gemm2b<<<N_NODES / 32, 224, 0, stream>>>(h, w2rT, b2, acc2, deg,
                                                (float*)d_out);
}
